// RNN_37005438222863
// MI455X (gfx1250) — matrix operand order repaired and checked by reading, not compiled, not run on hardware
//
#include <hip/hip_runtime.h>
#include <hip/hip_bf16.h>
#include <math.h>

// Model dims (fixed by the reference)
#define B_SZ   64
#define T_SZ   256
#define DD_SZ  256
#define H_SZ   512
#define LAT_SZ 32
#define DET_SZ 16

typedef __bf16 bf16_t;
typedef __attribute__((ext_vector_type(16))) __bf16 v16bf;
typedef __attribute__((ext_vector_type(8)))  float  v8f;

#define MBLK 2   // 16-row fragments per wave  -> 32 output rows
#define NBLK 4   // 16-col fragments per wave  -> 64 output cols

// ---------------------------------------------------------------------------
// Guard-free register-blocked WMMA GEMM:
//   C[M, NP] = act( A[M, Kp] @ W[NP, Kp]^T + bias )   (fp32 I/O, bf16 WMMA)
//
// Contracts (enforced by host-side padding):
//   * M  % 32 == 0, NP % 64 == 0
//   * Kp % 32 == 0, A row stride == Kp, pad columns of A are ZERO
//   * W is a zero-padded [NP, Kp] bf16 copy of the true [N, K] weights
//   * C row stride == NP; columns n >= N are written as 0 (keeps the next
//     GEMM's padded-K reads zero without a separate clear pass)
//
// One wave32 per 32x64 tile (2x4 fragments, 8 accumulators): per K-chunk the
// wave loads 4KB A + 4KB B and issues 8 v_wmma_f32_16x16x32_bf16
// (16 FLOP/byte vs 5.5 for an unblocked tile).
//
// Fragment layouts used here (wave32; A and B share a contiguous k order, not the ISA table order):
//   A 16x32 bf16 : lane m = lane&15; half = lane>>4
//                  a[0..7]  <- K = kk + half*16 + j
//                  a[8..15] <- K = kk + half*16 + 8 + j
//     -> two contiguous 32B float8 loads per fragment, cvt to bf16
//   B 32x16 bf16 : lane n = lane&15; b[j] <- K = kk + half*16 + j
//     -> one contiguous 32B v16bf load per fragment
//   C/D 16x16 f32: acc[r] -> row = r + 8*half, col = lane&15
// act: 0 = none, 1 = relu, 2 = tanh * scale
// ---------------------------------------------------------------------------
__global__ __launch_bounds__(32)
void wmma_gemm_bias_act(const float* __restrict__ A,
                        const bf16_t* __restrict__ W,
                        const float* __restrict__ bias,
                        float* __restrict__ C,
                        int M, int N, int NP, int Kp, int act, float scale)
{
    const int lane = threadIdx.x;
    const int half = lane >> 4;
    const int l15  = lane & 15;
    const int m0   = blockIdx.y * (16 * MBLK);
    const int n0   = blockIdx.x * (16 * NBLK);

    const float*  Arow = A + (size_t)(m0 + l15) * (size_t)Kp + half * 16;
    const bf16_t* Wrow = W + (size_t)(n0 + l15) * (size_t)Kp + half * 16;
    const size_t  AmStep = (size_t)16 * (size_t)Kp;   // next M fragment
    const size_t  WnStep = (size_t)16 * (size_t)Kp;   // next N fragment

    v8f acc[MBLK][NBLK] = {};

    for (int kk = 0; kk < Kp; kk += 32) {
        __builtin_prefetch(Arow + kk + 32, 0, 1);
        __builtin_prefetch(Wrow + kk + 32, 0, 1);

        // A fragments: two aligned 32B float8 loads each, convert to bf16
        v16bf afrag[MBLK];
        #pragma unroll
        for (int mb = 0; mb < MBLK; ++mb) {
            const float* ap = Arow + (size_t)mb * AmStep + kk;
            const v8f a0 = *(const v8f*)(ap);
            const v8f a1 = *(const v8f*)(ap + 8);
            #pragma unroll
            for (int j = 0; j < 8; ++j) {
                afrag[mb][j]     = (bf16_t)a0[j];
                afrag[mb][j + 8] = (bf16_t)a1[j];
            }
        }
        // B fragments: one aligned 32B load of 16 bf16 each
        v16bf bfrag[NBLK];
        #pragma unroll
        for (int nb = 0; nb < NBLK; ++nb)
            bfrag[nb] = *(const v16bf*)(Wrow + (size_t)nb * WnStep + kk);

        #pragma unroll
        for (int mb = 0; mb < MBLK; ++mb)
            #pragma unroll
            for (int nb = 0; nb < NBLK; ++nb)
                acc[mb][nb] = __builtin_amdgcn_wmma_f32_16x16x32_bf16(
                    false, afrag[mb], false, bfrag[nb],
                    (short)0, acc[mb][nb], false, false);
    }

    #pragma unroll
    for (int nb = 0; nb < NBLK; ++nb) {
        const int n = n0 + nb * 16 + l15;        // always < NP
        const bool nval = (n < N);
        const float bsum = (bias && nval) ? bias[n] : 0.f;
        #pragma unroll
        for (int mb = 0; mb < MBLK; ++mb) {
            #pragma unroll
            for (int r = 0; r < 8; ++r) {
                const int m = m0 + mb * 16 + r + 8 * half;   // always < M
                float v = acc[mb][nb][r] + bsum;
                if (act == 1)      v = fmaxf(v, 0.f);
                else if (act == 2) v = tanhf(v) * scale;
                C[(size_t)m * (size_t)NP + n] = nval ? v : 0.f;
            }
        }
    }
}

// ---------------------------------------------------------------------------
// Weight conversion with zero-padding: src [N,K] f32 -> dst [NP,Kp] bf16
// ---------------------------------------------------------------------------
__global__ void cvt_pad_kernel(const float* __restrict__ src,
                               bf16_t* __restrict__ dst,
                               int N, int K, int NP, int Kp)
{
    int i = blockIdx.x * blockDim.x + threadIdx.x;
    if (i >= NP * Kp) return;
    const int n = i / Kp;
    const int k = i - n * Kp;
    dst[i] = (n < N && k < K) ? (bf16_t)src[(size_t)n * K + k] : (bf16_t)0.f;
}

__global__ void zero_f32_kernel(float* __restrict__ p, int n)
{
    int i = blockIdx.x * blockDim.x + threadIdx.x;
    if (i < n) p[i] = 0.f;
}

// ---------------------------------------------------------------------------
// GRU gate math for one timestep t.
// gi_all: [B*T, 3H], row index b*T + t (x's native [B,T,D] order)
// gh:     [B, 3H]    h: [B, H] (updated in place)
// ---------------------------------------------------------------------------
__global__ void gru_gate_kernel(const float* __restrict__ gi_all,
                                const float* __restrict__ gh,
                                float* __restrict__ h, int t)
{
    int i = blockIdx.x * blockDim.x + threadIdx.x;  // over B*H
    if (i >= B_SZ * H_SZ) return;
    const int b = i / H_SZ;
    const int j = i - b * H_SZ;
    const float* gi  = gi_all + ((size_t)b * T_SZ + t) * (size_t)(3 * H_SZ);
    const float* ghb = gh + (size_t)b * (3 * H_SZ);
    const float ir = gi[j],              hr = ghb[j];
    const float iz = gi[H_SZ + j],       hz = ghb[H_SZ + j];
    const float in = gi[2 * H_SZ + j],   hn = ghb[2 * H_SZ + j];
    const float r = 1.f / (1.f + expf(-(ir + hr)));
    const float z = 1.f / (1.f + expf(-(iz + hz)));
    const float n = tanhf(in + r * hn);
    const float hv = h[i];
    h[i] = (1.f - z) * n + z * hv;
}

// m = mu + eps * exp(0.5 * logvar); mu/logvar have row stride 64 (padded NP)
__global__ void reparam_kernel(const float* __restrict__ mu,
                               const float* __restrict__ logvar,
                               const float* __restrict__ eps,
                               float* __restrict__ m)
{
    int i = blockIdx.x * blockDim.x + threadIdx.x;
    if (i >= B_SZ * LAT_SZ) return;
    const int b = i / LAT_SZ;
    const int j = i - b * LAT_SZ;
    const float muv = mu[b * 64 + j];
    const float lvv = logvar[b * 64 + j];
    m[i] = muv + eps[i] * expf(0.5f * lvv);
}

// ---------------------------------------------------------------------------
// Full latent scan in one block: z_t = z_{t-1} @ A[b]; z_seq[t,b,:] = z_t + b.
// dz/bvec have row stride 64 (padded NP). z_seq rows have stride 32 (K padded
// for the first decoder GEMM); pad columns [16,32) written as zero here.
// 1024 threads = B(64) x DET(16); LDS broadcast + barrier per step.
// ---------------------------------------------------------------------------
#define ZSTRIDE 32
__global__ __launch_bounds__(1024)
void latent_scan_kernel(const float* __restrict__ dz,
                        const float* __restrict__ bvec,
                        const float* __restrict__ Amat,   // [B, DET, DET]
                        float* __restrict__ z_seq)        // [T*B, ZSTRIDE]
{
    __shared__ float zs[B_SZ][DET_SZ];
    const int tid = threadIdx.x;
    const int b = tid / DET_SZ;
    const int e = tid - b * DET_SZ;

    float Acol[DET_SZ];                 // A[b, d, e] for all d
    #pragma unroll
    for (int d = 0; d < DET_SZ; ++d)
        Acol[d] = Amat[(size_t)b * DET_SZ * DET_SZ + d * DET_SZ + e];

    float z = dz[b * 64 + e];
    const float bias = bvec[b * 64 + e];
    z_seq[(size_t)b * ZSTRIDE + e] = z + bias;            // t = 0
    z_seq[(size_t)b * ZSTRIDE + DET_SZ + e] = 0.f;        // pad col

    for (int t = 1; t < T_SZ; ++t) {
        zs[b][e] = z;
        __syncthreads();
        float zn = 0.f;
        #pragma unroll
        for (int d = 0; d < DET_SZ; ++d)
            zn += zs[b][d] * Acol[d];
        __syncthreads();
        z = zn;
        const size_t row = (size_t)t * B_SZ + b;
        z_seq[row * ZSTRIDE + e] = z + bias;
        z_seq[row * ZSTRIDE + DET_SZ + e] = 0.f;
    }
}

// out[b,t,d] = (x[b,t,d] == 0) ? 0 : x_seq[t*B + b, d]
__global__ void mask_out_kernel(const float* __restrict__ x,
                                const float* __restrict__ x_seq,
                                float* __restrict__ out)
{
    size_t i = (size_t)blockIdx.x * blockDim.x + threadIdx.x;
    const size_t total = (size_t)B_SZ * T_SZ * DD_SZ;
    if (i >= total) return;
    const int d  = (int)(i % DD_SZ);
    const size_t bt = i / DD_SZ;
    const int t = (int)(bt % T_SZ);
    const int b = (int)(bt / T_SZ);
    const float xv = x[i];
    const float v  = x_seq[((size_t)t * B_SZ + b) * DD_SZ + d];
    out[i] = (xv == 0.f) ? 0.f : v;
}

// ---------------------------------------------------------------------------
// Host orchestration
// ---------------------------------------------------------------------------
static inline size_t align_up(size_t x, size_t a) { return (x + a - 1) & ~(a - 1); }

static void launch_gemm(hipStream_t s, const float* A, const bf16_t* W,
                        const float* bias, float* C,
                        int M, int N, int NP, int Kp, int act, float scale)
{
    dim3 grid(NP / (16 * NBLK), M / (16 * MBLK));
    wmma_gemm_bias_act<<<grid, 32, 0, s>>>(A, W, bias, C, M, N, NP, Kp, act, scale);
}

static void launch_cvt(hipStream_t s, const float* src, bf16_t* dst,
                       int N, int K, int NP, int Kp)
{
    int n = NP * Kp;
    cvt_pad_kernel<<<(n + 255) / 256, 256, 0, s>>>(src, dst, N, K, NP, Kp);
}

extern "C" void kernel_launch(void* const* d_in, const int* in_sizes, int n_in,
                              void* d_out, int out_size, void* d_ws, size_t ws_size,
                              hipStream_t stream)
{
    (void)in_sizes; (void)n_in; (void)out_size; (void)ws_size;

    // ---- inputs (setup_inputs dict order, params dict flattened in order) ----
    const float* x        = (const float*)d_in[0];   // [B,T,D]
    const float* eps      = (const float*)d_in[1];   // [B,LAT]
    const float* gru_Wih  = (const float*)d_in[2];   // [3H, D]
    const float* gru_bih  = (const float*)d_in[3];
    const float* gru_Whh  = (const float*)d_in[4];   // [3H, H]
    const float* gru_bhh  = (const float*)d_in[5];
    const float* menc1_w  = (const float*)d_in[6];   const float* menc1_b  = (const float*)d_in[7];
    const float* menc2_w  = (const float*)d_in[8];   const float* menc2_b  = (const float*)d_in[9];
    const float* menc31_w = (const float*)d_in[10];  const float* menc31_b = (const float*)d_in[11];
    const float* menc32_w = (const float*)d_in[12];  const float* menc32_b = (const float*)d_in[13];
    const float* aenc1_w  = (const float*)d_in[14];  const float* aenc1_b  = (const float*)d_in[15];
    const float* aenc2_w  = (const float*)d_in[16];  const float* aenc2_b  = (const float*)d_in[17];
    const float* aenc3A_w = (const float*)d_in[18];  const float* aenc3A_b = (const float*)d_in[19];
    const float* aenc3z_w = (const float*)d_in[20];  const float* aenc3z_b = (const float*)d_in[21];
    const float* aenc3b_w = (const float*)d_in[22];  const float* aenc3b_b = (const float*)d_in[23];
    const float* dec1_w   = (const float*)d_in[24];  const float* dec1_b   = (const float*)d_in[25];
    const float* dec2_w   = (const float*)d_in[26];  const float* dec2_b   = (const float*)d_in[27];
    const float* dec3_w   = (const float*)d_in[28];  const float* dec3_b   = (const float*)d_in[29];
    const float* dec4_w   = (const float*)d_in[30];  const float* dec4_b   = (const float*)d_in[31];
    // d_in[32] = max_length (== T, compile-time constant here)

    // ---- workspace bump allocator ----
    char* ws = (char*)d_ws;
    size_t off = 0;
    auto alloc = [&](size_t bytes) -> char* {
        char* p = ws + off;
        off = align_up(off + bytes, 256);
        return p;
    };

    // Padded bf16 weight copies: [NP, Kp], zero-filled pads (~11 MB, L2-resident)
    bf16_t* wih_bf    = (bf16_t*)alloc((size_t)1536 * 256  * 2);
    bf16_t* whh_bf    = (bf16_t*)alloc((size_t)1536 * 512  * 2);
    bf16_t* menc1_bf  = (bf16_t*)alloc((size_t)1024 * 512  * 2);
    bf16_t* menc2_bf  = (bf16_t*)alloc((size_t)512  * 1024 * 2);
    bf16_t* menc31_bf = (bf16_t*)alloc((size_t)64   * 512  * 2);
    bf16_t* menc32_bf = (bf16_t*)alloc((size_t)64   * 512  * 2);
    bf16_t* aenc1_bf  = (bf16_t*)alloc((size_t)1024 * 32   * 2);
    bf16_t* aenc2_bf  = (bf16_t*)alloc((size_t)1024 * 1024 * 2);
    bf16_t* aenc3A_bf = (bf16_t*)alloc((size_t)256  * 1024 * 2);
    bf16_t* aenc3z_bf = (bf16_t*)alloc((size_t)64   * 1024 * 2);
    bf16_t* aenc3b_bf = (bf16_t*)alloc((size_t)64   * 1024 * 2);
    bf16_t* dec1_bf   = (bf16_t*)alloc((size_t)512  * 32   * 2);
    bf16_t* dec2_bf   = (bf16_t*)alloc((size_t)1024 * 512  * 2);
    bf16_t* dec3_bf   = (bf16_t*)alloc((size_t)1024 * 1024 * 2);
    bf16_t* dec4_bf   = (bf16_t*)alloc((size_t)256  * 1024 * 2);

    // fp32 activation buffers (strides = producing GEMM's NP)
    float* h      = (float*)alloc((size_t)B_SZ * 512 * 4);
    float* gh     = (float*)alloc((size_t)B_SZ * 1536 * 4);
    float* h1     = (float*)alloc((size_t)B_SZ * 1024 * 4);   // N=1000, NP=1024
    float* h2     = (float*)alloc((size_t)B_SZ * 512 * 4);    // N=500,  NP=512
    float* mu     = (float*)alloc((size_t)B_SZ * 64 * 4);     // N=32,   NP=64
    float* logvar = (float*)alloc((size_t)B_SZ * 64 * 4);     // N=32,   NP=64
    float* m      = (float*)alloc((size_t)B_SZ * 32 * 4);     // dense [B,32]
    float* a1     = (float*)alloc((size_t)B_SZ * 1024 * 4);   // N=1000, NP=1024
    float* a2     = (float*)alloc((size_t)B_SZ * 1024 * 4);   // N=1000, NP=1024
    float* Amat   = (float*)alloc((size_t)B_SZ * 256 * 4);
    float* dz     = (float*)alloc((size_t)B_SZ * 64 * 4);     // N=16,   NP=64
    float* bb     = (float*)alloc((size_t)B_SZ * 64 * 4);     // N=16,   NP=64
    float* z_seq  = (float*)alloc((size_t)T_SZ * B_SZ * ZSTRIDE * 4);
    // big regions with reuse:
    //   big1: gi_all (16384x1536) -> d2 (16384x1024) -> x_seq (16384x256)
    //   big2: d1 (16384x512)      -> d3 (16384x1024)
    const size_t MROWS = (size_t)B_SZ * T_SZ;                 // 16384
    float* big1 = (float*)alloc(MROWS * 1536 * 4);
    float* big2 = (float*)alloc(MROWS * 1024 * 4);
    float* gi_all = big1;
    float* d1 = big2;
    float* d2 = big1;
    float* d3 = big2;
    float* x_seq = big1;

    // ---- stage 0: weights fp32 -> zero-padded bf16 [NP, Kp] ----
    launch_cvt(stream, gru_Wih,  wih_bf,    1536, 256,  1536, 256);
    launch_cvt(stream, gru_Whh,  whh_bf,    1536, 512,  1536, 512);
    launch_cvt(stream, menc1_w,  menc1_bf,  1000, 512,  1024, 512);
    launch_cvt(stream, menc2_w,  menc2_bf,  500,  1000, 512,  1024);
    launch_cvt(stream, menc31_w, menc31_bf, 32,   500,  64,   512);
    launch_cvt(stream, menc32_w, menc32_bf, 32,   500,  64,   512);
    launch_cvt(stream, aenc1_w,  aenc1_bf,  1000, 32,   1024, 32);
    launch_cvt(stream, aenc2_w,  aenc2_bf,  1000, 1000, 1024, 1024);
    launch_cvt(stream, aenc3A_w, aenc3A_bf, 256,  1000, 256,  1024);
    launch_cvt(stream, aenc3z_w, aenc3z_bf, 16,   1000, 64,   1024);
    launch_cvt(stream, aenc3b_w, aenc3b_bf, 16,   1000, 64,   1024);
    launch_cvt(stream, dec1_w,   dec1_bf,   500,  16,   512,  32);
    launch_cvt(stream, dec2_w,   dec2_bf,   1000, 500,  1024, 512);
    launch_cvt(stream, dec3_w,   dec3_bf,   1000, 1000, 1024, 1024);
    launch_cvt(stream, dec4_w,   dec4_bf,   256,  1000, 256,  1024);

    // ---- stage 1: hoisted input projection  gi_all = X @ W_ih^T + b_ih ----
    // x viewed as [B*T, 256]; row r = b*T + t (native layout, no transpose).
    launch_gemm(stream, x, wih_bf, gru_bih, gi_all,
                (int)MROWS, 1536, 1536, 256, 0, 1.f);

    // ---- stage 2: sequential GRU scan (only h @ W_hh^T per step) ----
    zero_f32_kernel<<<(B_SZ * H_SZ + 255) / 256, 256, 0, stream>>>(h, B_SZ * H_SZ);
    for (int t = 0; t < T_SZ; ++t) {
        launch_gemm(stream, h, whh_bf, gru_bhh, gh, B_SZ, 1536, 1536, 512, 0, 1.f);
        gru_gate_kernel<<<(B_SZ * H_SZ + 255) / 256, 256, 0, stream>>>(gi_all, gh, h, t);
    }

    // ---- stage 3: encoder_m + reparameterize ----
    launch_gemm(stream, h,  menc1_bf,  menc1_b,  h1,     B_SZ, 1000, 1024, 512,  1, 1.f);
    launch_gemm(stream, h1, menc2_bf,  menc2_b,  h2,     B_SZ, 500,  512,  1024, 1, 1.f);
    launch_gemm(stream, h2, menc31_bf, menc31_b, mu,     B_SZ, 32,   64,   512,  0, 1.f);
    launch_gemm(stream, h2, menc32_bf, menc32_b, logvar, B_SZ, 32,   64,   512,  0, 1.f);
    reparam_kernel<<<(B_SZ * LAT_SZ + 255) / 256, 256, 0, stream>>>(mu, logvar, eps, m);

    // ---- stage 4: encoder_Azb ----
    launch_gemm(stream, m,  aenc1_bf,  aenc1_b,  a1,   B_SZ, 1000, 1024, 32,   1, 1.f);
    launch_gemm(stream, a1, aenc2_bf,  aenc2_b,  a2,   B_SZ, 1000, 1024, 1024, 1, 1.f);
    launch_gemm(stream, a2, aenc3A_bf, aenc3A_b, Amat, B_SZ, 256,  256,  1024, 2, 1.f / DET_SZ);
    launch_gemm(stream, a2, aenc3z_bf, aenc3z_b, dz,   B_SZ, 16,   64,   1024, 0, 1.f);
    launch_gemm(stream, a2, aenc3b_bf, aenc3b_b, bb,   B_SZ, 16,   64,   1024, 0, 1.f);

    // ---- stage 5: latent linear dynamics (whole 256-step scan, one block) ----
    latent_scan_kernel<<<1, B_SZ * DET_SZ, 0, stream>>>(dz, bb, Amat, z_seq);

    // ---- stage 6: decoder over all T*B rows ----
    launch_gemm(stream, z_seq, dec1_bf, dec1_b, d1,    (int)MROWS, 500,  512,  32,   1, 1.f);
    launch_gemm(stream, d1,    dec2_bf, dec2_b, d2,    (int)MROWS, 1000, 1024, 512,  1, 1.f);
    launch_gemm(stream, d2,    dec3_bf, dec3_b, d3,    (int)MROWS, 1000, 1024, 1024, 1, 1.f);
    launch_gemm(stream, d3,    dec4_bf, dec4_b, x_seq, (int)MROWS, 256,  256,  1024, 0, 1.f);

    // ---- stage 7: mask + [T,B,D] -> [B,T,D] ----
    {
        const size_t total = (size_t)B_SZ * T_SZ * DD_SZ;
        mask_out_kernel<<<(unsigned)((total + 255) / 256), 256, 0, stream>>>(
            x, x_seq, (float*)d_out);
    }
}
